// RelGraphEncoderTG_54047868453344
// MI455X (gfx1250) — matrix operand order repaired and checked by reading, not compiled, not run on hardware
//
#include <hip/hip_runtime.h>
#include <math.h>

#define NNODES 50000
#define DFEAT  128
#define NREL   16
#define NBASE  16
#define NEDGE  1600000
#define KREL   (NREL * DFEAT)        // 2048
#define KTOT   (KREL + DFEAT)        // 2176 (relations + root block)
#define NTILES (KTOT / 32)           // 68
#define LNEPS  1e-5f

typedef __attribute__((ext_vector_type(16))) __bf16 v16bf;
typedef __attribute__((ext_vector_type(8)))  float  v8f;

#if defined(__has_builtin)
#if __has_builtin(__builtin_amdgcn_cvt_pk_bf16_f32)
#define HAVE_PK_BF16 1
#endif
#endif

__device__ __forceinline__ unsigned int f2bf(float f) {
    unsigned int u = __builtin_bit_cast(unsigned int, f);
    u += 0x7FFFu + ((u >> 16) & 1u);          // round-to-nearest-even
    return u >> 16;
}

// pack two floats into two bf16 (low|high)
__device__ __forceinline__ unsigned int pk_bf16(float x, float y) {
#ifdef HAVE_PK_BF16
    typedef __attribute__((ext_vector_type(2))) __bf16 v2bf;
    v2bf p = __builtin_amdgcn_cvt_pk_bf16_f32(x, y);   // v_cvt_pk_bf16_f32
    return __builtin_bit_cast(unsigned int, p);
#else
    // round-to-nearest (ties away) + single v_perm_b32 pack: 3 VALU total
    unsigned int ux = __builtin_bit_cast(unsigned int, x) + 0x8000u;
    unsigned int uy = __builtin_bit_cast(unsigned int, y) + 0x8000u;
    // result bytes = { uy[3], uy[2], ux[3], ux[2] }  (S0=uy -> idx 4..7, S1=ux -> idx 0..3)
    return __builtin_amdgcn_perm(uy, ux, 0x07060302u);
#endif
}

// ---------------------------------------------------------------- init h ----
__global__ void init_h_kernel(const float* __restrict__ x,
                              const unsigned char* __restrict__ mask,
                              float* __restrict__ h) {
    int i = blockIdx.x * blockDim.x + threadIdx.x;
    if (i < NNODES * DFEAT)
        h[i] = mask[i >> 7] ? x[i] : 0.0f;
}

// ------------------------------------------- fused weights, transposed bf16 -
// wallT[f][k], k<2048: W[r,d,f] = sum_b comp[r,b]*bases[b,d,f]; k>=2048: root[d,f]
__global__ void build_wallT_kernel(const float* __restrict__ bases,
                                   const float* __restrict__ comp,
                                   const float* __restrict__ root,
                                   unsigned short* __restrict__ wallT) {
    int i = blockIdx.x * blockDim.x + threadIdx.x;
    if (i >= DFEAT * KTOT) return;
    int f = i / KTOT;
    int k = i - f * KTOT;
    float v;
    if (k < KREL) {
        int r = k >> 7, d = k & 127;
        float acc = 0.0f;
#pragma unroll
        for (int b = 0; b < NBASE; ++b)
            acc = fmaf(comp[r * NBASE + b], bases[(b * DFEAT + d) * DFEAT + f], acc);
        v = acc;
    } else {
        v = root[(k - KREL) * DFEAT + f];
    }
    wallT[(size_t)f * KTOT + k] = (unsigned short)f2bf(v);
}

// ------------------------------------------------------------- scatter ------
// one wave32 per edge; lane handles 4 consecutive features (float4 gather)
__global__ void scatter_kernel(const float* __restrict__ h,
                               const int* __restrict__ ei,
                               const int* __restrict__ et,
                               float* __restrict__ agg,
                               float* __restrict__ cnt) {
    long long t = (long long)blockIdx.x * blockDim.x + threadIdx.x;
    int e    = (int)(t >> 5);
    int lane = (int)(t & 31);
    if (e >= NEDGE) return;
    int src = ei[e];
    int dst = ei[NEDGE + e];
    int r   = et[e];
    int seg = dst * NREL + r;
    float4 hv = ((const float4*)(h + (size_t)src * DFEAT))[lane];
    float* a = agg + (size_t)seg * DFEAT + lane * 4;
    unsafeAtomicAdd(a + 0, hv.x);
    unsafeAtomicAdd(a + 1, hv.y);
    unsafeAtomicAdd(a + 2, hv.z);
    unsafeAtomicAdd(a + 3, hv.w);
    if (lane == 0) unsafeAtomicAdd(cnt + seg, 1.0f);
}

// -------------------------------------------- fused GEMM+GELU+residual+LN ---
// block = 256 threads = 8 waves; block owns 16 rows (nodes), wave owns 16 cols.
// A tile staged in LDS directly in WMMA A-fragment order (bf16 pairs),
// double-buffered: one barrier per K-tile, global latency hidden behind WMMA.
__launch_bounds__(256)
__global__ void rgcn_fused_kernel(const float* __restrict__ h_in,
                                  const float* __restrict__ agg,
                                  const float* __restrict__ cnt,
                                  const unsigned short* __restrict__ wallT,
                                  const float* __restrict__ bias,
                                  const float* __restrict__ lnw,
                                  const float* __restrict__ lnb,
                                  const unsigned char* __restrict__ mask,
                                  int mask_out,
                                  float* __restrict__ out) {
    __shared__ unsigned int afrag_s[2][32][8]; // [buf][dest lane][bf16 pair]
    __shared__ float sc_s[16][NREL];           // 1/max(cnt,1) per row, per rel
    __shared__ float lnbuf[16][DFEAT];         // pre-LN values

    const int tid  = threadIdx.x;
    const int lane = tid & 31;
    const int wave = tid >> 5;
    const int n0   = blockIdx.x * 16;
    const int m    = lane & 15;
    const int half = lane >> 4;
    const int col  = wave * 16 + m;            // output feature owned by lane

    // ---- one-time: per-row, per-relation mean scales (v_rcp, not IEEE div)
    {
        const int mm = tid >> 4, r = tid & 15;
        const float c = cnt[(n0 + mm) * NREL + r];
        sc_s[mm][r] = __builtin_amdgcn_rcpf(fmaxf(c, 1.0f));
    }

    // ---- loader-thread geometry: thread -> (dest lane, pair q) -> (row, k)
    const int Ld  = tid >> 3;                  // destination lane 0..31
    const int qd  = tid & 7;                   // bf16-pair index 0..7
    const int md  = Ld & 15;                   // source row in tile
    const int kbd = (Ld >> 4) * 16;            // A-frag K-group select
    const int kkd = kbd + 2 * qd;
    const float* aggrow = agg  + (size_t)(n0 + md) * KREL  + kkd;
    const float* hrow   = h_in + (size_t)(n0 + md) * DFEAT + kkd;

    auto gload = [&](int t) -> float2 {
        const int k0 = t * 32;
        return (k0 < KREL) ? *(const float2*)(aggrow + k0)
                           : *(const float2*)(hrow + (k0 - KREL));
    };
    auto sstore = [&](float2 g, int t, int b) {
        const int k0 = t * 32;
        const float s = (k0 < KREL) ? sc_s[md][k0 >> 7] : 1.0f;
        afrag_s[b][Ld][qd] = pk_bf16(g.x * s, g.y * s);
    };

    v8f acc = {};

    // prologue: tile 0 -> buf 0
    __syncthreads();                           // sc_s ready
    sstore(gload(0), 0, 0);
    __syncthreads();

    for (int t = 0; t < NTILES; ++t) {
        const int cur = t & 1;
        float2 gn;
        if (t + 1 < NTILES) gn = gload(t + 1); // issue next tile's loads early

        // consume: A fragment = 8 contiguous uints per lane (2x ds_load_b128)
        union { v16bf v; unsigned int u[8]; } af;
#pragma unroll
        for (int q = 0; q < 8; ++q) af.u[q] = afrag_s[cur][lane][q];
        const v16bf bfrag =
            *(const v16bf*)(wallT + (size_t)col * KTOT + t * 32 + half * 16);

        acc = __builtin_amdgcn_wmma_f32_16x16x32_bf16(false, af.v, false, bfrag,
                                                      (short)0, acc, false, false);

        if (t + 1 < NTILES) sstore(gn, t + 1, cur ^ 1);
        __syncthreads();
    }

    // ---- epilogue: bias, exact GELU, residual -> LDS (C layout: M = i+8*half)
#pragma unroll
    for (int i = 0; i < 8; ++i) {
        const int row = i + half * 8;
        float v = acc[i] + bias[col];
        float g = 0.5f * v * (1.0f + erff(v * 0.70710678118654752f));
        float res = h_in[(size_t)(n0 + row) * DFEAT + col];
        lnbuf[row][col] = res + g;
    }
    __syncthreads();

    // ---- LayerNorm: each wave reduces rows {wave, wave+8} over 128 feats
#pragma unroll
    for (int rr = 0; rr < 2; ++rr) {
        const int row = wave + rr * 8;
        const int n   = n0 + row;
        const int f   = lane * 4;
        const float4 vv = *(const float4*)&lnbuf[row][f];
        float s  = vv.x + vv.y + vv.z + vv.w;
        float sq = vv.x * vv.x + vv.y * vv.y + vv.z * vv.z + vv.w * vv.w;
#pragma unroll
        for (int o = 16; o >= 1; o >>= 1) {
            s  += __shfl_xor(s,  o, 32);
            sq += __shfl_xor(sq, o, 32);
        }
        const float mu  = s * (1.0f / DFEAT);
        const float var = sq * (1.0f / DFEAT) - mu * mu;
        const float inv = __builtin_amdgcn_rsqf(var + LNEPS);
        const bool keep = (!mask_out) || (mask[n] != 0);
        float4 o4;
        o4.x = keep ? (vv.x - mu) * inv * lnw[f + 0] + lnb[f + 0] : 0.0f;
        o4.y = keep ? (vv.y - mu) * inv * lnw[f + 1] + lnb[f + 1] : 0.0f;
        o4.z = keep ? (vv.z - mu) * inv * lnw[f + 2] + lnb[f + 2] : 0.0f;
        o4.w = keep ? (vv.w - mu) * inv * lnw[f + 3] + lnb[f + 3] : 0.0f;
        *(float4*)(out + (size_t)n * DFEAT + f) = o4;
    }
}

// ---------------------------------------------------------------------------
extern "C" void kernel_launch(void* const* d_in, const int* in_sizes, int n_in,
                              void* d_out, int out_size, void* d_ws, size_t ws_size,
                              hipStream_t stream) {
    const float*         x   = (const float*)d_in[0];
    const int*           ei  = (const int*)d_in[1];
    const int*           et  = (const int*)d_in[2];
    const unsigned char* msk = (const unsigned char*)d_in[3];

    char* w = (char*)d_ws;
    float* agg = (float*)w;  w += (size_t)NNODES * NREL * DFEAT * sizeof(float);
    float* cnt = (float*)w;  w += (size_t)NNODES * NREL * sizeof(float);
    float* h0  = (float*)w;  w += (size_t)NNODES * DFEAT * sizeof(float);
    float* h1  = (float*)w;  w += (size_t)NNODES * DFEAT * sizeof(float);
    unsigned short* wallT = (unsigned short*)w;

    {
        const int total = NNODES * DFEAT;
        init_h_kernel<<<(total + 255) / 256, 256, 0, stream>>>(x, msk, h0);
    }

    const float* hcur = h0;
    for (int l = 0; l < 2; ++l) {
        const float* bases = (const float*)d_in[4 + 6 * l];
        const float* comp  = (const float*)d_in[5 + 6 * l];
        const float* root  = (const float*)d_in[6 + 6 * l];
        const float* bias  = (const float*)d_in[7 + 6 * l];
        const float* lnw   = (const float*)d_in[8 + 6 * l];
        const float* lnb   = (const float*)d_in[9 + 6 * l];
        float* outp = (l == 0) ? h1 : (float*)d_out;

        // agg and cnt are contiguous: one async memset
        hipMemsetAsync(agg, 0, (size_t)NNODES * NREL * (DFEAT + 1) * sizeof(float), stream);

        build_wallT_kernel<<<(DFEAT * KTOT + 255) / 256, 256, 0, stream>>>(bases, comp, root, wallT);

        {
            const long long tot = (long long)NEDGE * 32;
            scatter_kernel<<<(int)((tot + 255) / 256), 256, 0, stream>>>(hcur, ei, et, agg, cnt);
        }

        rgcn_fused_kernel<<<NNODES / 16, 256, 0, stream>>>(hcur, agg, cnt, wallT, bias,
                                                           lnw, lnb, msk,
                                                           (l == 1) ? 1 : 0, outp);
        hcur = outp;
    }
}
